// RNN_79903571575179
// MI455X (gfx1250) — compile-verified, matrix operand order checked by reading, not run on hardware
//
#include <hip/hip_runtime.h>

// RNN: out[0]=h; out[t]=tanh(x[t]@Wx + out[t-1]@Wh + bh), t=1..255
//  Phase 1 (parallel): out[g,:] = x_flat[g,:]@Wx + bh via one 65536x1024x512 WMMA GEMM
//  Phase 2 (sequential): 255 steps out[t]=tanh(out[t] + h_{t-1}@Wh); h kept in bf16
//  ping-pong buffers so the critical path has zero f32->bf16 conversion VALU.
//  B panels staged in LDS via CDNA5 async global->LDS loads (ASYNCcnt); B fragments
//  batch-loaded from LDS ahead of each WMMA burst so ds latency overlaps compute.

#define LATENT 1024
#define INPUT  512
#define SEQ    256
#define BATCH  256

typedef __attribute__((ext_vector_type(16))) __bf16        v16bf;
typedef __attribute__((ext_vector_type(2)))  __bf16        v2bf;
typedef __attribute__((ext_vector_type(8)))  float         v8f;
typedef __attribute__((ext_vector_type(4)))  float         v4f;
typedef __attribute__((ext_vector_type(4)))  unsigned int  v4u;

// Trivially-constructible fragment (all ext_vector members -> union is legal)
union FragBF {
  v16bf          v;
  v4u            q[2];
  unsigned int   w[8];
  unsigned short u[16];
};

// fp32 -> bf16 RNE (bit pattern)
static __device__ __forceinline__ unsigned short f2bf(float f) {
  unsigned int x = __float_as_uint(f);
  unsigned int r = (x + 0x7FFFu + ((x >> 16) & 1u)) >> 16;
  return (unsigned short)r;
}

// two fp32 -> packed bf16x2 (one v_cvt_pk_bf16_f32 when available)
static __device__ __forceinline__ unsigned int pk2bf(float a, float b) {
#if __has_builtin(__builtin_amdgcn_cvt_pk_bf16_f32)
  v2bf p = __builtin_amdgcn_cvt_pk_bf16_f32(a, b);
  return __builtin_bit_cast(unsigned int, p);
#else
  return (unsigned)f2bf(a) | ((unsigned)f2bf(b) << 16);
#endif
}

// ---------------------------------------------------------------------------
// Generic block GEMM: C[8 waves x MTILES*16 rows x 64 cols] += A @ B,
// B given as BT[N][K] bf16 (N-major -> contiguous 16B runs per lane).
// B panel staged in LDS with async global->LDS copies; per K-step all four
// B fragments are batch-loaded into registers before the WMMA burst.
// Fragment layouts per CDNA5 ISA 7.12.2.
// ---------------------------------------------------------------------------
template<int KSIZE, int MTILES, bool A_BF16, bool ADD_BIAS, bool RECURRENT, bool WRITE_H>
__global__ void __launch_bounds__(256)
k_gemm(const float* __restrict__ Af,            // A fp32 (if !A_BF16)
       const unsigned short* __restrict__ Abf,  // A bf16 (if A_BF16), row-major [M][K]
       const unsigned short* __restrict__ BT,   // [N][K] bf16 bits
       const float* __restrict__ bias,
       float* __restrict__ C,                   // [M][LATENT]; RECURRENT: pre-act in/out
       unsigned short* __restrict__ Hout)       // bf16 copy of result (if WRITE_H)
{
  __shared__ unsigned short Bs[64 * KSIZE];     // 64 cols x K panel (<=128 KB)

  const int lane = threadIdx.x & 31;
  const int wave = threadIdx.x >> 5;
  const int n0 = blockIdx.x * 64;
  const int m0 = (blockIdx.y * 8 + wave) * (MTILES * 16);

  // ---- stage B panel: contiguous 64*K bf16 block, async global -> LDS ----
  {
    const unsigned short* __restrict__ src = BT + (size_t)n0 * KSIZE;
    const unsigned lbase = (unsigned)(size_t)(&Bs[0]);   // LDS offset = low 32 bits
    constexpr int CH = (64 * KSIZE) / (8 * 256);         // 16B per thread per chunk
    #pragma unroll
    for (int c = 0; c < CH; ++c) {
      const int e = (c * 256 + threadIdx.x) * 8;         // ushort index
      const unsigned loff = lbase + e * 2;
      const unsigned short* g = src + e;
      asm volatile("global_load_async_to_lds_b128 %0, %1, off"
                   :: "v"(loff), "v"(g) : "memory");
    }
    asm volatile("s_wait_asynccnt 0x0" ::: "memory");
    __syncthreads();
  }

  v8f acc[MTILES][4];
  #pragma unroll
  for (int mt = 0; mt < MTILES; ++mt)
    #pragma unroll
    for (int j = 0; j < 4; ++j)
      #pragma unroll
      for (int r = 0; r < 8; ++r) acc[mt][j][r] = 0.0f;

  const int kbase = (lane >> 4) * 8;      // A/B lane K-run base
  const int mrow0 = m0 + (lane & 15);     // A lane row
  const int nloc  = (lane & 15);          // B lane col within panel

  #pragma unroll 2
  for (int k0 = 0; k0 < KSIZE; k0 += 32) {
    // ---- A fragments (per M tile): two contiguous 16B runs per lane ----
    FragBF af[MTILES];
    #pragma unroll
    for (int mt = 0; mt < MTILES; ++mt) {
      const int row = mrow0 + mt * 16;
      if (A_BF16) {
        const unsigned short* ar = Abf + (size_t)row * KSIZE;
        af[mt].q[0] = *(const v4u*)(ar + k0 + kbase);
        af[mt].q[1] = *(const v4u*)(ar + k0 + kbase + 16);
      } else {
        const float* ar = Af + (size_t)row * KSIZE;
        v4f f0 = *(const v4f*)(ar + k0 + kbase);
        v4f f1 = *(const v4f*)(ar + k0 + kbase + 4);
        v4f f2 = *(const v4f*)(ar + k0 + kbase + 16);
        v4f f3 = *(const v4f*)(ar + k0 + kbase + 20);
        af[mt].w[0] = pk2bf(f0[0], f0[1]); af[mt].w[1] = pk2bf(f0[2], f0[3]);
        af[mt].w[2] = pk2bf(f1[0], f1[1]); af[mt].w[3] = pk2bf(f1[2], f1[3]);
        af[mt].w[4] = pk2bf(f2[0], f2[1]); af[mt].w[5] = pk2bf(f2[2], f2[3]);
        af[mt].w[6] = pk2bf(f3[0], f3[1]); af[mt].w[7] = pk2bf(f3[2], f3[3]);
      }
    }
    // ---- batch-load all 4 B fragments from LDS (one ds clause, waits overlap) ----
    FragBF barr[4];
    #pragma unroll
    for (int j = 0; j < 4; ++j) {
      const int bidx = (j * 16 + nloc) * KSIZE + k0 + kbase;
      barr[j].q[0] = *(const v4u*)(&Bs[bidx]);
      barr[j].q[1] = *(const v4u*)(&Bs[bidx + 16]);
    }
    // ---- WMMA burst: each B fragment reused for all M tiles ----
    #pragma unroll
    for (int j = 0; j < 4; ++j) {
      #pragma unroll
      for (int mt = 0; mt < MTILES; ++mt) {
        acc[mt][j] = __builtin_amdgcn_wmma_f32_16x16x32_bf16(
            false, af[mt].v, false, barr[j].v, (short)0, acc[mt][j], false, false);
      }
    }
  }

  // ---- epilogue: C lane map M = r + 8*(lane/16), N = lane%16 ----
  #pragma unroll
  for (int mt = 0; mt < MTILES; ++mt) {
    #pragma unroll
    for (int j = 0; j < 4; ++j) {
      const int ncol = n0 + j * 16 + nloc;
      const float b = ADD_BIAS ? bias[ncol] : 0.0f;
      #pragma unroll
      for (int r = 0; r < 8; ++r) {
        const int row = m0 + mt * 16 + r + 8 * (lane >> 4);
        float* cp = C + (size_t)row * LATENT + ncol;
        float val = acc[mt][j][r] + b;
        if (RECURRENT) val = tanhf(val + *cp);
        *cp = val;
        if (WRITE_H) Hout[(size_t)row * LATENT + ncol] = f2bf(val);
      }
    }
  }
}

// W[K][N] fp32 -> WT[N][K] bf16 bits (N-major: B fragments become contiguous)
__global__ void k_t2bf(const float* __restrict__ W, unsigned short* __restrict__ WT,
                       int K, int N) {
  int idx = blockIdx.x * blockDim.x + threadIdx.x;
  if (idx >= K * N) return;
  int k = idx / N, n = idx - k * N;
  WT[(size_t)n * K + k] = f2bf(W[idx]);
}

// x fp32 -> bf16, 8 elements/thread (16B stores)
__global__ void k_x2bf(const float* __restrict__ X, unsigned short* __restrict__ Xbf,
                       int n8) {
  int i = blockIdx.x * blockDim.x + threadIdx.x;
  if (i >= n8) return;
  const v4f* p = (const v4f*)(X + (size_t)i * 8);
  v4f a = p[0], b = p[1];
  v4u o;
  o[0] = pk2bf(a[0], a[1]); o[1] = pk2bf(a[2], a[3]);
  o[2] = pk2bf(b[0], b[1]); o[3] = pk2bf(b[2], b[3]);
  *(v4u*)(Xbf + (size_t)i * 8) = o;
}

// out[0] = h (fp32) and Hbf0 = bf16(h)
__global__ void k_init(const float* __restrict__ h, float* __restrict__ dst,
                       unsigned short* __restrict__ hbf, int n) {
  int i = blockIdx.x * blockDim.x + threadIdx.x;
  if (i >= n) return;
  float v = h[i];
  dst[i] = v;
  hbf[i] = f2bf(v);
}

extern "C" void kernel_launch(void* const* d_in, const int* in_sizes, int n_in,
                              void* d_out, int out_size, void* d_ws, size_t ws_size,
                              hipStream_t stream) {
  const float* x  = (const float*)d_in[0];   // [SEQ, BATCH, INPUT]
  const float* h  = (const float*)d_in[1];   // [BATCH, LATENT]
  const float* Wx = (const float*)d_in[2];   // [INPUT, LATENT]
  const float* Wh = (const float*)d_in[3];   // [LATENT, LATENT]
  const float* bh = (const float*)d_in[4];   // [LATENT]
  float* out = (float*)d_out;                // [SEQ, BATCH, LATENT]

  // ws layout (ushort elements)
  unsigned short* WxT  = (unsigned short*)d_ws;                       // 1 MB
  unsigned short* WhT  = WxT  + (size_t)LATENT * INPUT;               // 2 MB
  unsigned short* Hbf0 = WhT  + (size_t)LATENT * LATENT;              // 0.5 MB
  unsigned short* Hbf1 = Hbf0 + (size_t)BATCH * LATENT;               // 0.5 MB
  unsigned short* Xbf  = Hbf1 + (size_t)BATCH * LATENT;               // 64 MB (optional)
  const size_t need_xbf = ((size_t)LATENT * INPUT + (size_t)LATENT * LATENT +
                           2u * (size_t)BATCH * LATENT + (size_t)SEQ * BATCH * INPUT) *
                          sizeof(unsigned short);
  const bool big_ws = ws_size >= need_xbf;

  // Phase 0: weight conversion (once), bf16 N-major
  k_t2bf<<<(INPUT * LATENT + 255) / 256, 256, 0, stream>>>(Wx, WxT, INPUT, LATENT);
  k_t2bf<<<(LATENT * LATENT + 255) / 256, 256, 0, stream>>>(Wh, WhT, LATENT, LATENT);

  // Phase 1: pre-activations for all timesteps (parallel GEMM, 2 M-tiles/wave)
  dim3 g1(LATENT / 64, (SEQ * BATCH) / 256);   // (16, 256)
  if (big_ws) {
    const int n8 = SEQ * BATCH * INPUT / 8;
    k_x2bf<<<(n8 + 255) / 256, 256, 0, stream>>>(x, Xbf, n8);
    k_gemm<INPUT, 2, true, true, false, false><<<g1, 256, 0, stream>>>(
        nullptr, Xbf, WxT, bh, out, nullptr);
  } else {
    k_gemm<INPUT, 2, false, true, false, false><<<g1, 256, 0, stream>>>(
        x, nullptr, WxT, bh, out, nullptr);
  }

  // out[0] = h, and seed the bf16 hidden-state ping-pong buffer
  k_init<<<(BATCH * LATENT + 255) / 256, 256, 0, stream>>>(h, out, Hbf0, BATCH * LATENT);

  // Phase 2: 255 sequential recurrence GEMMs; A is bf16 (no conversion VALU on
  // the critical path), Wh panel restaged from L2 into LDS each step.
  dim3 g2(LATENT / 64, (BATCH / 16) / 8);      // (16, 2)
  for (int t = 1; t < SEQ; ++t) {
    unsigned short* Hprev = (t & 1) ? Hbf0 : Hbf1;
    unsigned short* Hnext = (t & 1) ? Hbf1 : Hbf0;
    k_gemm<LATENT, 1, true, false, true, true><<<g2, 256, 0, stream>>>(
        nullptr, Hprev, WhT, nullptr, out + (size_t)t * BATCH * LATENT, Hnext);
  }
}
